// UFOAttention_56521769615656
// MI455X (gfx1250) — matrix operand order repaired and checked by reading, not compiled, not run on hardware
//
#include <hip/hip_runtime.h>

#define BATCH 8
#define HEADS 8
#define HD    64
#define EMBED 512
#define SEQL  8192
#define NCHUNK 4

typedef __attribute__((ext_vector_type(16))) __bf16 v16bf;
typedef __attribute__((ext_vector_type(8)))  float  v8f;

union TileU { v16bf v; uint4 q[2]; __bf16 h[16]; };

// A operand 16x32 (MxK) bf16 from row-major [stride] memory.
// ISA layout: lanes 0-15 row=lane, K={kb..kb+7, kb+16..kb+23}; lanes 16-31 kb+=8.
__device__ __forceinline__ v16bf load_a16x32_bf(const __bf16* base, int stride, int kk) {
  const int lane = threadIdx.x & 31;
  const int row  = lane & 15;
  const int kb   = kk + ((lane >> 4) << 3);
  const __bf16* p = base + (size_t)row * stride + kb;
  TileU u;
  u.q[0] = *reinterpret_cast<const uint4*>(p);
  u.q[1] = *reinterpret_cast<const uint4*>(p + 16);
  return u.v;
}

// B operand 32x16 (KxN) where B = T^t and T is row-major [N rows, stride].
// ISA layout: lane n holds column n; lanes 0-15 K={kk..kk+7, kk+16..kk+23}; lanes 16-31 kb+=8.
__device__ __forceinline__ v16bf load_b32x16_T(const __bf16* baseT, int stride, int kk) {
  const int lane = threadIdx.x & 31;
  const int col  = lane & 15;
  const int kb   = kk + ((lane >> 4) << 3);
  const __bf16* p = baseT + (size_t)col * stride + kb;
  TileU u;
  u.q[0] = *reinterpret_cast<const uint4*>(p);
  u.q[1] = *reinterpret_cast<const uint4*>(p + 16);
  return u.v;
}

// A operand from fp32 source with native bf16 conversion (v_cvt_pk_bf16_f32).
__device__ __forceinline__ v16bf load_a16x32_f32(const float* base, int stride, int kk) {
  const int lane = threadIdx.x & 31;
  const int row  = lane & 15;
  const int kb   = kk + ((lane >> 4) << 3);
  const float* p = base + (size_t)row * stride + kb;
  float4 f0 = *reinterpret_cast<const float4*>(p);
  float4 f1 = *reinterpret_cast<const float4*>(p + 4);
  float4 f2 = *reinterpret_cast<const float4*>(p + 16);
  float4 f3 = *reinterpret_cast<const float4*>(p + 20);
  TileU u;
  u.h[0]  = (__bf16)f0.x; u.h[1]  = (__bf16)f0.y; u.h[2]  = (__bf16)f0.z; u.h[3]  = (__bf16)f0.w;
  u.h[4]  = (__bf16)f1.x; u.h[5]  = (__bf16)f1.y; u.h[6]  = (__bf16)f1.z; u.h[7]  = (__bf16)f1.w;
  u.h[8]  = (__bf16)f2.x; u.h[9]  = (__bf16)f2.y; u.h[10] = (__bf16)f2.z; u.h[11] = (__bf16)f2.w;
  u.h[12] = (__bf16)f3.x; u.h[13] = (__bf16)f3.y; u.h[14] = (__bf16)f3.z; u.h[15] = (__bf16)f3.w;
  return u.v;
}

#define WMMA_BF16(a, b, c) \
  __builtin_amdgcn_wmma_f32_16x16x32_bf16(false, (a), false, (b), (short)0, (c), false, false)

// 0) Convert weights fp32 -> bf16 once per call.
__global__ void cvt_weights_kernel(const float* __restrict__ Wq, const float* __restrict__ Wk,
                                   const float* __restrict__ Wv, const float* __restrict__ Wo,
                                   __bf16* __restrict__ WqB, __bf16* __restrict__ WkB,
                                   __bf16* __restrict__ WvB, __bf16* __restrict__ WoB) {
  const int i = blockIdx.x * blockDim.x + threadIdx.x;
  if (i < HD * HD) { WqB[i] = (__bf16)Wq[i]; WkB[i] = (__bf16)Wk[i]; WvB[i] = (__bf16)Wv[i]; }
  if (i < EMBED * EMBED) WoB[i] = (__bf16)Wo[i];
}

// 1) Per-head projections + in-register L4 norm for q.
__global__ __launch_bounds__(128) void proj_kernel(
    const float* __restrict__ Q, const float* __restrict__ K, const float* __restrict__ V,
    const __bf16* __restrict__ WqB, const __bf16* __restrict__ WkB,
    const __bf16* __restrict__ WvB, const float* __restrict__ gamma,
    __bf16* __restrict__ Qn, __bf16* __restrict__ KpT, __bf16* __restrict__ VpT) {
  const int wid     = (blockIdx.x * blockDim.x + threadIdx.x) >> 5;
  const int mtiles  = SEQL / 16;
  const int per_mat = BATCH * HEADS * mtiles;
  const int mat = wid / per_mat;
  const int rem = wid - mat * per_mat;
  const int bh  = rem / mtiles;
  const int mt  = rem - bh * mtiles;
  const int b = bh >> 3, h = bh & 7;
  const int l0 = mt * 16;

  const float* src = (mat == 0) ? Q : (mat == 1) ? K : V;
  const __bf16* Wb = (mat == 0) ? WqB : (mat == 1) ? WkB : WvB;
  const float* abase = src + ((size_t)b * SEQL + l0) * EMBED + h * HD;

  v16bf a0 = load_a16x32_f32(abase, EMBED, 0);
  v16bf a1 = load_a16x32_f32(abase, EMBED, 32);

  v8f acc[4];
#pragma unroll
  for (int nt = 0; nt < 4; ++nt) {
    v8f c = {};
    c = WMMA_BF16(a0, load_b32x16_T(Wb + nt * 16 * HD, HD, 0), c);
    c = WMMA_BF16(a1, load_b32x16_T(Wb + nt * 16 * HD, HD, 32), c);
    acc[nt] = c;
  }

  const int lane = threadIdx.x & 31;
  const int n = lane & 15, hi = lane >> 4;

  if (mat == 0) {
    // Row m = rr + 8*hi; its 64 e-values live in acc[0..3][rr] across the 16
    // lanes of this half. Reduce x^4 within the half (masks 1..8 stay in-half).
    float p[8];
#pragma unroll
    for (int rr = 0; rr < 8; ++rr) {
      float s = 0.f;
#pragma unroll
      for (int nt = 0; nt < 4; ++nt) { float x = acc[nt][rr]; float x2 = x * x; s += x2 * x2; }
#pragma unroll
      for (int m = 1; m < 16; m <<= 1) s += __shfl_xor(s, m, 32);
      p[rr] = s;
    }
    const float g = gamma[h];
#pragma unroll
    for (int rr = 0; rr < 8; ++rr) {
      const float sc = g / sqrtf(sqrtf(p[rr]));
      const int l = l0 + rr + 8 * hi;
#pragma unroll
      for (int nt = 0; nt < 4; ++nt)
        Qn[((size_t)bh * SEQL + l) * HD + nt * 16 + n] = (__bf16)(acc[nt][rr] * sc);
    }
  } else {
    __bf16* dstT = (mat == 1) ? KpT : VpT;
#pragma unroll
    for (int nt = 0; nt < 4; ++nt)
#pragma unroll
      for (int rr = 0; rr < 8; ++rr) {
        const int l = l0 + rr + 8 * hi;
        const int d = nt * 16 + n;
        dstT[((size_t)bh * HD + d) * SEQL + l] = (__bf16)acc[nt][rr];
      }
  }
}

// 2) kv partials over L (deterministic, no atomics). Dual accumulator chains for ILP.
__global__ __launch_bounds__(128) void kv_partial_kernel(
    const __bf16* __restrict__ KpT, const __bf16* __restrict__ VpT,
    float* __restrict__ kvp) {
  const int wid = (blockIdx.x * blockDim.x + threadIdx.x) >> 5;
  const int bh  = wid >> 6;
  const int rem = wid & 63;
  const int tile = rem >> 2, c = rem & 3;
  const int mt = tile >> 2, nt = tile & 3;
  const __bf16* Ab = KpT + ((size_t)bh * HD + mt * 16) * SEQL;
  const __bf16* Bb = VpT + ((size_t)bh * HD + nt * 16) * SEQL;
  const int CH = SEQL / NCHUNK;
  v8f acc0 = {}, acc1 = {};
  for (int kk = c * CH; kk < (c + 1) * CH; kk += 64) {
    __builtin_prefetch(Ab + (size_t)(threadIdx.x & 31) * SEQL + kk + 256, 0, 0);
    acc0 = WMMA_BF16(load_a16x32_bf(Ab, SEQL, kk),      load_b32x16_T(Bb, SEQL, kk),      acc0);
    acc1 = WMMA_BF16(load_a16x32_bf(Ab, SEQL, kk + 32), load_b32x16_T(Bb, SEQL, kk + 32), acc1);
  }
  v8f acc = acc0 + acc1;
  const int lane = threadIdx.x & 31;
  const int n = lane & 15, hi = lane >> 4;
  float* dst = kvp + ((size_t)c * (BATCH * HEADS) + bh) * (HD * HD);
#pragma unroll
  for (int rr = 0; rr < 8; ++rr)
    dst[(mt * 16 + rr + 8 * hi) * HD + nt * 16 + n] = acc[rr];
}

// 3) Reduce partials, L4-norm over e, scale by gamma, store TRANSPOSED kvnT[e][d].
__global__ __launch_bounds__(128) void kv_norm_kernel(
    const float* __restrict__ kvp, const float* __restrict__ gamma,
    __bf16* __restrict__ kvnT) {
  const int row  = (blockIdx.x * blockDim.x + threadIdx.x) >> 5;  // bh*64 + d
  const int lane = threadIdx.x & 31;
  const int bh = row >> 6, d = row & 63, h = bh & 7;
  float x0 = 0.f, x1 = 0.f;
#pragma unroll
  for (int c = 0; c < NCHUNK; ++c) {
    const float* src = kvp + ((size_t)c * (BATCH * HEADS) + bh) * (HD * HD) + d * HD;
    x0 += src[lane];
    x1 += src[lane + 32];
  }
  float s = x0 * x0 * x0 * x0 + x1 * x1 * x1 * x1;
#pragma unroll
  for (int m = 1; m < 32; m <<= 1) s += __shfl_xor(s, m, 32);
  const float sc = gamma[h] / sqrtf(sqrtf(s));
  __bf16* dst = kvnT + (size_t)bh * (HD * HD);
  dst[lane * HD + d]        = (__bf16)(x0 * sc);
  dst[(lane + 32) * HD + d] = (__bf16)(x1 * sc);
}

// 4) Out = q_n @ kv_n; store head-concat Oh [B, L, H*D] bf16.
__global__ __launch_bounds__(128) void attn_out_kernel(
    const __bf16* __restrict__ Qn, const __bf16* __restrict__ kvnT,
    __bf16* __restrict__ Oh) {
  const int wid    = (blockIdx.x * blockDim.x + threadIdx.x) >> 5;
  const int mtiles = SEQL / 16;
  const int bh = wid / mtiles;
  const int mt = wid - bh * mtiles;
  const int b = bh >> 3, h = bh & 7;
  const __bf16* Ab = Qn + ((size_t)bh * SEQL + mt * 16) * HD;
  v16bf a0 = load_a16x32_bf(Ab, HD, 0);
  v16bf a1 = load_a16x32_bf(Ab, HD, 32);
  const __bf16* Bt = kvnT + (size_t)bh * (HD * HD);
  const int lane = threadIdx.x & 31;
  const int n = lane & 15, hi = lane >> 4;
#pragma unroll
  for (int nt = 0; nt < 4; ++nt) {
    v8f c = {};
    c = WMMA_BF16(a0, load_b32x16_T(Bt + nt * 16 * HD, HD, 0), c);
    c = WMMA_BF16(a1, load_b32x16_T(Bt + nt * 16 * HD, HD, 32), c);
#pragma unroll
    for (int rr = 0; rr < 8; ++rr) {
      const int l = mt * 16 + rr + 8 * hi;
      Oh[((size_t)b * SEQL + l) * EMBED + h * HD + nt * 16 + n] = (__bf16)c[rr];
    }
  }
}

// 5) Final projection + bias. Two N-tiles per wave: A loaded once, 2x WMMA ILP.
__global__ __launch_bounds__(128) void final_kernel(
    const __bf16* __restrict__ Oh, const __bf16* __restrict__ WoB,
    const float* __restrict__ bo, float* __restrict__ out) {
  const int wid = (blockIdx.x * blockDim.x + threadIdx.x) >> 5;
  const int mt  = wid >> 4;   // 4096 M-tiles over B*L
  const int ntp = wid & 15;   // 16 N-tile pairs over 512
  const __bf16* Ab  = Oh  + (size_t)mt * 16 * EMBED;
  const __bf16* Bt0 = WoB + (size_t)(2 * ntp) * 16 * EMBED;
  const __bf16* Bt1 = Bt0 + 16 * EMBED;
  v8f acc0 = {}, acc1 = {};
#pragma unroll
  for (int kk = 0; kk < EMBED; kk += 32) {
    __builtin_prefetch(Ab + (size_t)(threadIdx.x & 31) * EMBED + kk + 128, 0, 0);
    v16bf a = load_a16x32_bf(Ab, EMBED, kk);
    acc0 = WMMA_BF16(a, load_b32x16_T(Bt0, EMBED, kk), acc0);
    acc1 = WMMA_BF16(a, load_b32x16_T(Bt1, EMBED, kk), acc1);
  }
  const int lane = threadIdx.x & 31;
  const int n = lane & 15, hi = lane >> 4;
  const float bias0 = bo[2 * ntp * 16 + n];
  const float bias1 = bo[(2 * ntp + 1) * 16 + n];
#pragma unroll
  for (int rr = 0; rr < 8; ++rr) {
    const size_t r = (size_t)mt * 16 + rr + 8 * hi;
    out[r * EMBED + 2 * ntp * 16 + n]       = acc0[rr] + bias0;
    out[r * EMBED + (2 * ntp + 1) * 16 + n] = acc1[rr] + bias1;
  }
}

extern "C" void kernel_launch(void* const* d_in, const int* in_sizes, int n_in,
                              void* d_out, int out_size, void* d_ws, size_t ws_size,
                              hipStream_t stream) {
  const float* Q     = (const float*)d_in[0];
  const float* K     = (const float*)d_in[1];
  const float* V     = (const float*)d_in[2];
  const float* Wq    = (const float*)d_in[3];
  const float* Wk    = (const float*)d_in[4];
  const float* Wv    = (const float*)d_in[5];
  const float* Wo    = (const float*)d_in[6];
  const float* bo    = (const float*)d_in[7];
  const float* gamma = (const float*)d_in[8];
  float* out = (float*)d_out;

  char* w = (char*)d_ws;
  const size_t NQ    = (size_t)BATCH * HEADS * SEQL * HD;        // 33,554,432 elems
  const size_t KVPSZ = (size_t)NCHUNK * BATCH * HEADS * HD * HD;  // 1,048,576 f32
  __bf16* Qn   = (__bf16*)(w + NQ * 0);
  __bf16* KpT  = (__bf16*)(w + NQ * 2);
  __bf16* VpT  = (__bf16*)(w + NQ * 4);
  __bf16* Oh   = (__bf16*)(w + NQ * 6);
  float*  kvp  = (float*)(w + NQ * 8);
  __bf16* WqB  = (__bf16*)(w + NQ * 8 + KVPSZ * sizeof(float));
  __bf16* WkB  = WqB + HD * HD;
  __bf16* WvB  = WkB + HD * HD;
  __bf16* WoB  = WvB + HD * HD;
  __bf16* kvnT = WoB + EMBED * EMBED;

  cvt_weights_kernel<<<(EMBED * EMBED + 255) / 256, 256, 0, stream>>>(
      Wq, Wk, Wv, Wo, WqB, WkB, WvB, WoB);

  proj_kernel<<<(3 * BATCH * HEADS * (SEQL / 16)) / 4, 128, 0, stream>>>(
      Q, K, V, WqB, WkB, WvB, gamma, Qn, KpT, VpT);

  kv_partial_kernel<<<(BATCH * HEADS * 16 * NCHUNK) / 4, 128, 0, stream>>>(KpT, VpT, kvp);

  kv_norm_kernel<<<(BATCH * HEADS * HD) / 4, 128, 0, stream>>>(kvp, gamma, kvnT);

  attn_out_kernel<<<(BATCH * HEADS * (SEQL / 16)) / 4, 128, 0, stream>>>(Qn, kvnT, Oh);

  final_kernel<<<((BATCH * SEQL / 16) * (EMBED / 32)) / 4, 128, 0, stream>>>(Oh, WoB, bo, out);
}
